// VectorQuantizer_42339787604562
// MI455X (gfx1250) — matrix operand order repaired and checked by reading, not compiled, not run on hardware
//
#include <hip/hip_runtime.h>
#include <hip/hip_bf16.h>

typedef __attribute__((ext_vector_type(16))) _Float16     v16h;
typedef __attribute__((ext_vector_type(8)))  float        v8f;
typedef __attribute__((ext_vector_type(8)))  unsigned int v8u;

#define NUM_EMB   512
#define DIM       64
#define THW       65536          // T*H*W = 16*64*64
#define N_TOTAL   524288         // B*T*H*W
#define Q_ELEMS   4194304        // B*C*T*H*W
#define EMB_STR   72             // padded LDS stride (halves): 144 B rows, 16B-aligned
#define PASSES    4              // 4 x 128 rows per block
#define COMMIT    0.025f

union Frag { v8u u; v16h h; };

// ---- CDNA5 async global->LDS copy (ASYNCcnt), 16 bytes per lane ----
__device__ __forceinline__ void async_copy16(unsigned lds_off, const void* gaddr) {
    asm volatile("global_load_async_to_lds_b128 %0, %1, off"
                 :: "v"(lds_off), "v"((unsigned long long)(uintptr_t)gaddr)
                 : "memory");
}
__device__ __forceinline__ void wait_async0() {
    asm volatile("s_wait_asynccnt 0x0" ::: "memory");
}
// low 32 bits of a generic LDS pointer == LDS byte offset (ISA 10.2 aperture map)
__device__ __forceinline__ unsigned lds_off(const void* p) {
    return (unsigned)(uintptr_t)p;
}

// ---------------------------------------------------------------------------
// Kernel 0: one-time codebook prep: f32 -> f16 into ws, plus ||e||^2. Also
// zeroes the loss slot (stream order guarantees visibility downstream).
// ---------------------------------------------------------------------------
__global__ __launch_bounds__(256) void vq_prep_kernel(
    const float* __restrict__ emb, _Float16* __restrict__ wsEmb,
    float* __restrict__ wsE2, float* __restrict__ lossSlot)
{
    const int i = blockIdx.x * 256 + threadIdx.x;   // 32768 threads
    if (i == 0) *lossSlot = 0.0f;
    wsEmb[i] = (_Float16)emb[i];
    if (i < NUM_EMB) {
        const float* er = emb + i * DIM;
        float s = 0.0f;
#pragma unroll
        for (int c = 0; c < DIM; ++c) { float v = er[c]; s += v * v; }
        wsE2[i] = s;
    }
}

// ---------------------------------------------------------------------------
// Kernel 1: f16 WMMA distance GEMM + argmin. One block = 512 rows (4 passes
// of 128), 8 waves; codebook staged once per block via async-to-LDS; A tiles
// double-buffered and software-pipelined with async copies.
// ---------------------------------------------------------------------------
__global__ __launch_bounds__(256) void vq_argmin_kernel(
    const float* __restrict__ x, const _Float16* __restrict__ wsEmb,
    const float* __restrict__ wsE2, int* __restrict__ outIdx)
{
    __shared__ __align__(16) _Float16 sEmb[NUM_EMB * EMB_STR];   // 73,728 B
    __shared__ __align__(16) float    sE2[NUM_EMB];              //  2,048 B
    __shared__ __align__(16) float    sAf[2][DIM * 128];         // 2x32,768 B, [c][row]

    const int t = threadIdx.x;
    const int n0 = blockIdx.x * 512;      // 512 | THW -> block stays in one b
    const int b = n0 >> 16;
    const int thw_blk = n0 & 65535;
    const char* xB = (const char*)(x + (size_t)b * DIM * THW);

    // --- async-stage f16 codebook: 512 rows x 128 B = 4096 16B chunks ---
    {
        const unsigned embL = lds_off(&sEmb[0]);
        const char* eg = (const char*)wsEmb;
#pragma unroll
        for (int i = 0; i < 16; ++i) {
            int ci = t + 256 * i;
            int code = ci >> 3, part = ci & 7;
            async_copy16(embL + code * (EMB_STR * 2) + part * 16,
                         eg + code * 128 + part * 16);
        }
    }
    // --- async-stage A pass tile: 64 ch x 128 rows f32 = 2048 16B chunks ---
    auto issueA = [&](int p, int bufSel) {
        const unsigned aL = lds_off(&sAf[bufSel][0]);
        const int thw0 = thw_blk + p * 128;
#pragma unroll
        for (int i = 0; i < 8; ++i) {
            int ci = t + 256 * i;
            int c = ci >> 5, part = ci & 31;          // [c][row] chunk order
            async_copy16(aL + (unsigned)(c * 128 + part * 4) * 4,
                         xB + ((size_t)c * THW + thw0 + part * 4) * 4);
        }
    };
    issueA(0, 0);
    for (int code = t; code < NUM_EMB; code += 256) sE2[code] = wsE2[code];
    wait_async0();
    __syncthreads();

    const int lane = t & 31, wave = t >> 5;
    const int m    = lane & 15;           // A row within wave strip
    const int half = lane >> 4;           // wave32 K-half select
    const int rloc = wave * 16 + m;       // row within 128-row pass
    const int n     = m;                  // B column within 16-code tile
    const int kbase = half * 8;
    const int base  = half * 8;

    for (int p = 0; p < PASSES; ++p) {
        if (p + 1 < PASSES) issueA(p + 1, (p + 1) & 1);   // overlap with compute

        const float* Af = &sAf[p & 1][0];
        // A fragments (16-bit 16x32 layout), f32->f16 convert at build time
        Frag fa0, fa1;
#pragma unroll
        for (int v = 0; v < 8; ++v) {
            int kk = (v < 4) ? (base + 2 * v) : (8 + base + 2 * v);
            fa0.h[2 * v]     = (_Float16)Af[kk * 128 + rloc];
            fa0.h[2 * v + 1] = (_Float16)Af[(kk + 1) * 128 + rloc];
            fa1.h[2 * v]     = (_Float16)Af[(32 + kk) * 128 + rloc];
            fa1.h[2 * v + 1] = (_Float16)Af[(33 + kk) * 128 + rloc];
        }

        float bestS[8];
        int   bestI[8];
#pragma unroll
        for (int r = 0; r < 8; ++r) { bestS[r] = 3.0e38f; bestI[r] = 0; }

#pragma unroll 2
        for (int j = 0; j < 32; ++j) {    // 32 tiles of 16 codes
            const int code = j * 16 + n;
            const unsigned* Er = (const unsigned*)(sEmb + code * EMB_STR + kbase);
            Frag fb0, fb1;
#pragma unroll
            for (int v = 0; v < 8; ++v) { const int w = (v < 4) ? v : v + 4; fb0.u[v] = Er[w]; fb1.u[v] = Er[16 + w]; }

            v8f acc = {};
            acc = __builtin_amdgcn_wmma_f32_16x16x32_f16(false, fa0.h, false, fb0.h,
                                                         (short)0, acc, false, false);
            acc = __builtin_amdgcn_wmma_f32_16x16x32_f16(false, fa1.h, false, fb1.h,
                                                         (short)0, acc, false, false);
            const float e2 = sE2[code];
#pragma unroll
            for (int r = 0; r < 8; ++r) { // row M = r + 8*half (C/D layout)
                float s = e2 - 2.0f * acc[r];
                if (s < bestS[r]) { bestS[r] = s; bestI[r] = code; }
            }
        }

        // reduce best across the 16 lanes (16 column subsets) of each half
#pragma unroll
        for (int mask = 1; mask <= 8; mask <<= 1) {
#pragma unroll
            for (int r = 0; r < 8; ++r) {
                float os = __shfl_xor(bestS[r], mask, 32);
                int   oi = __shfl_xor(bestI[r], mask, 32);
                if (os < bestS[r] || (os == bestS[r] && oi < bestI[r])) {
                    bestS[r] = os; bestI[r] = oi;
                }
            }
        }
        if ((lane & 15) == 0) {
#pragma unroll
            for (int r = 0; r < 8; ++r)
                outIdx[n0 + p * 128 + wave * 16 + half * 8 + r] = bestI[r];
        }

        wait_async0();       // next A tile landed (copied during compute)
        __syncthreads();     // all waves done with this pass + see next tile
    }
}

// ---------------------------------------------------------------------------
// Kernel 2: gather codes -> transposed output [B,C,T,H,W], accumulate loss.
// ---------------------------------------------------------------------------
__global__ __launch_bounds__(256) void vq_gather_kernel(
    const float* __restrict__ x, const float* __restrict__ emb,
    const int* __restrict__ idx, float* __restrict__ q,
    float* __restrict__ lossSlot)
{
    __shared__ float red[256];
    const int t = threadIdx.x;
    const size_t o = (size_t)blockIdx.x * 256 + t;
    const int thw = (int)(o & 65535);
    const int bc  = (int)(o >> 16);
    const int c   = bc & 63;
    const int b   = bc >> 6;
    const int n   = b * THW + thw;

    const int   k  = idx[n];
    const float e  = emb[k * DIM + c];     // 128 KB table: stays in L2/WGP$
    const float xv = x[o];
    q[o] = e;                              // straight-through value == quantized
    const float d = e - xv;
    red[t] = d * d;
    __syncthreads();
    for (int s = 128; s > 0; s >>= 1) {
        if (t < s) red[t] += red[t + s];
        __syncthreads();
    }
    if (t == 0)
        atomicAdd(lossSlot, red[0] * ((1.0f + COMMIT) / (float)Q_ELEMS));
}

// ---------------------------------------------------------------------------
// Kernel 3: indices -> float region of d_out.
// ---------------------------------------------------------------------------
__global__ __launch_bounds__(256) void vq_idx_kernel(
    const int* __restrict__ idx, float* __restrict__ out)
{
    const int n = blockIdx.x * 256 + threadIdx.x;
    out[n] = (float)idx[n];
}

// ---------------------------------------------------------------------------
extern "C" void kernel_launch(void* const* d_in, const int* in_sizes, int n_in,
                              void* d_out, int out_size, void* d_ws, size_t ws_size,
                              hipStream_t stream) {
    const float* x   = (const float*)d_in[0];   // [8,64,16,64,64] f32
    const float* emb = (const float*)d_in[1];   // [512,64] f32
    float* out = (float*)d_out;                 // quantized | loss | indices(f32)
    float* q        = out;
    float* lossSlot = out + Q_ELEMS;
    float* idxOut   = out + Q_ELEMS + 1;

    // workspace: f16 codebook (64 KB) | e2 (2 KB) | indices (2 MB)
    _Float16* wsEmb = (_Float16*)d_ws;
    float*    wsE2  = (float*)((char*)d_ws + 65536);
    int*      wsIdx = (int*)((char*)d_ws + 65536 + 2048);

    vq_prep_kernel  <<<NUM_EMB * DIM / 256, 256, 0, stream>>>(emb, wsEmb, wsE2, lossSlot);
    vq_argmin_kernel<<<N_TOTAL / 512,       256, 0, stream>>>(x, wsEmb, wsE2, wsIdx);
    vq_gather_kernel<<<Q_ELEMS / 256,       256, 0, stream>>>(x, emb, wsIdx, q, lossSlot);
    vq_idx_kernel   <<<N_TOTAL / 256,       256, 0, stream>>>(wsIdx, idxOut);
}
